// GNNComponent_54391465837039
// MI455X (gfx1250) — matrix operand order repaired and checked by reading, not compiled, not run on hardware
//
#include <hip/hip_runtime.h>
#include <hip/hip_bf16.h>
#include <math.h>

typedef __attribute__((ext_vector_type(16))) _Float16 v16h;
typedef __attribute__((ext_vector_type(8)))  float    v8f;

#define NEG_SLOPE 0.2f
#define LN_EPS    1e-5f

// ---------------------------------------------------------------------------
// GEMM: out[M x 64] = A[M x K] @ W[K x 64], f16 WMMA with fp32 accumulation.
// One wave computes a 16x64 output tile (4 accumulators, reusing the A frag).
// W is staged in LDS transposed with +8 padding so each B fragment is a
// contiguous 32B run per lane (vectorizable ds loads, banks spread).
// ---------------------------------------------------------------------------
__global__ void gat_gemm_wmma(const float* __restrict__ A,
                              const float* __restrict__ Wm,
                              float* __restrict__ out,
                              int M, int K) {
  const int KP = K + 8;                       // padded LDS row stride
  __shared__ _Float16 Wt[64 * (128 + 8)];     // max K=128 -> 17 KB

  const int tid = threadIdx.x;
  // Stage W transposed: Wt[n*KP + k] = W[k*64 + n]
  for (int i = tid; i < 64 * K; i += blockDim.x) {
    int n = i / K, k = i - n * K;
    Wt[n * KP + k] = (_Float16)Wm[k * 64 + n];
  }
  __syncthreads();

  const int lane   = tid & 31;
  const int wave   = tid >> 5;
  const int tile   = blockIdx.x * (blockDim.x >> 5) + wave;
  const int tilesM = (M + 15) >> 4;
  if (tile >= tilesM) return;                 // uniform per wave: EXEC stays full

  const int m    = lane & 15;
  const int row  = tile * 16 + m;
  const int rowc = row < M ? row : M - 1;     // clamp (M=100000 is exact anyway)
  const int kbA  = (lane >> 4) * 16;          // A frag: lanes 16-31 hold K+16..31
  const int kbB  = (lane >> 4) * 16;          // B frag: lanes 16-31 hold K+16..31
  const int ncol = lane & 15;

  v8f c0 = {}, c1 = {}, c2 = {}, c3 = {};
  for (int k0 = 0; k0 < K; k0 += 32) {
    // A fragment: contiguous layout (element s: K=kb+s)
    v16h a;
    const float* ap = A + (size_t)rowc * K + k0 + kbA;
#pragma unroll
    for (int hh = 0; hh < 2; ++hh)
#pragma unroll
      for (int j = 0; j < 8; ++j)
        a[8 * hh + j] = (_Float16)ap[8 * hh + j];

    // B fragments: 16 contiguous f16 per lane from transposed LDS image
    v16h b0, b1, b2, b3;
    const _Float16* w0 = &Wt[(ncol +  0) * KP + k0 + kbB];
    const _Float16* w1 = &Wt[(ncol + 16) * KP + k0 + kbB];
    const _Float16* w2 = &Wt[(ncol + 32) * KP + k0 + kbB];
    const _Float16* w3 = &Wt[(ncol + 48) * KP + k0 + kbB];
#pragma unroll
    for (int e = 0; e < 16; ++e) {
      b0[e] = w0[e]; b1[e] = w1[e]; b2[e] = w2[e]; b3[e] = w3[e];
    }

    c0 = __builtin_amdgcn_wmma_f32_16x16x32_f16(false, a, false, b0, (short)0, c0, false, false);
    c1 = __builtin_amdgcn_wmma_f32_16x16x32_f16(false, a, false, b1, (short)0, c1, false, false);
    c2 = __builtin_amdgcn_wmma_f32_16x16x32_f16(false, a, false, b2, (short)0, c2, false, false);
    c3 = __builtin_amdgcn_wmma_f32_16x16x32_f16(false, a, false, b3, (short)0, c3, false, false);
  }

  // C layout: lanes 0-15 -> M=r, lanes 16-31 -> M=r+8, N=lane&15
  const int mBase = tile * 16 + ((lane >> 4) ? 8 : 0);
#pragma unroll
  for (int r = 0; r < 8; ++r) {
    int orow = mBase + r;
    if (orow < M) {
      float* op = out + (size_t)orow * 64 + ncol;
      op[0]  = c0[r];
      op[16] = c1[r];
      op[32] = c2[r];
      op[48] = c3[r];
    }
  }
}

// ---------------------------------------------------------------------------
// Per-node attention dots: sS[i] = h[i,:]·aS, sD[i] = h[i,:]·aD  (wave/node)
// ---------------------------------------------------------------------------
__global__ void gat_node_dots(const float* __restrict__ h,
                              const float* __restrict__ aS,
                              const float* __restrict__ aD,
                              float* __restrict__ sS, float* __restrict__ sD,
                              int N) {
  int wv = blockIdx.x * (blockDim.x >> 5) + (threadIdx.x >> 5);
  int lane = threadIdx.x & 31;
  if (wv >= N) return;
  const float* hp = h + (size_t)wv * 64;
  float v0 = hp[lane], v1 = hp[lane + 32];
  float d0 = v0 * aS[lane] + v1 * aS[lane + 32];
  float d1 = v0 * aD[lane] + v1 * aD[lane + 32];
#pragma unroll
  for (int m = 16; m; m >>= 1) {
    d0 += __shfl_xor(d0, m, 32);
    d1 += __shfl_xor(d1, m, 32);
  }
  if (lane == 0) { sS[wv] = d0; sD[wv] = d1; }
}

// ---------------------------------------------------------------------------
// Reset: agg = 0 (N*64), maxv = -inf, denom = 0 (N)
// ---------------------------------------------------------------------------
__global__ void gat_reset(float* __restrict__ agg, float* __restrict__ maxv,
                          float* __restrict__ denom, int N) {
  int i = blockIdx.x * blockDim.x + threadIdx.x;
  if (i < N * 64) agg[i] = 0.f;
  if (i < N) {
    maxv[i]  = __int_as_float((int)0xff800000u);  // -inf
    denom[i] = 0.f;
  }
}

__device__ __forceinline__ void edge_sd(const int* __restrict__ ei, int t,
                                        int E, int& s, int& d) {
  if (t < E) { s = ei[t]; d = ei[E + t]; }
  else       { s = t - E; d = t - E; }          // appended self-loops
}
__device__ __forceinline__ float lrelu(float x) {
  return x > 0.f ? x : NEG_SLOPE * x;
}
__device__ __forceinline__ void fatomic_max(float* p, float v) {
  if (v >= 0.f) atomicMax((int*)p, __float_as_int(v));
  else          atomicMin((unsigned int*)p, __float_as_uint(v));
}
__device__ __forceinline__ void fatomic_add(float* p, float v) {
  __hip_atomic_fetch_add(p, v, __ATOMIC_RELAXED, __HIP_MEMORY_SCOPE_AGENT);
}

// ---------------------------------------------------------------------------
// Segment softmax pass 1: running max per destination node
// ---------------------------------------------------------------------------
__global__ void gat_edge_max(const int* __restrict__ ei,
                             const float* __restrict__ sS,
                             const float* __restrict__ sD,
                             float* __restrict__ maxv, int E, int ET) {
  int t = blockIdx.x * blockDim.x + threadIdx.x;
  if (t >= ET) return;
  int s, d; edge_sd(ei, t, E, s, d);
  fatomic_max(maxv + d, lrelu(sS[s] + sD[d]));
}

// ---------------------------------------------------------------------------
// Segment softmax pass 2: denom[d] += exp(e - max[d])
// ---------------------------------------------------------------------------
__global__ void gat_edge_expsum(const int* __restrict__ ei,
                                const float* __restrict__ sS,
                                const float* __restrict__ sD,
                                const float* __restrict__ maxv,
                                float* __restrict__ denom, int E, int ET) {
  int t = blockIdx.x * blockDim.x + threadIdx.x;
  if (t >= ET) return;
  int s, d; edge_sd(ei, t, E, s, d);
  float e = lrelu(sS[s] + sD[d]);
  fatomic_add(denom + d, __expf(e - maxv[d]));
}

// ---------------------------------------------------------------------------
// Aggregation: wave per edge; agg[d,:] += alpha * h[s,:]  (2 feats/lane,
// coalesced 128B rows; atomics land in L2 since agg fits in 192MB L2)
// ---------------------------------------------------------------------------
__global__ void gat_edge_aggregate(const int* __restrict__ ei,
                                   const float* __restrict__ sS,
                                   const float* __restrict__ sD,
                                   const float* __restrict__ maxv,
                                   const float* __restrict__ denom,
                                   const float* __restrict__ h,
                                   float* __restrict__ agg, int E, int ET) {
  int wv = blockIdx.x * (blockDim.x >> 5) + (threadIdx.x >> 5);
  if (wv >= ET) return;
  int lane = threadIdx.x & 31;
  int s, d; edge_sd(ei, wv, E, s, d);
  float e = lrelu(sS[s] + sD[d]);
  float alpha = __expf(e - maxv[d]) / denom[d];
  const float* hp = h + (size_t)s * 64;
  float* op = agg + (size_t)d * 64;
  fatomic_add(op + lane,      hp[lane]      * alpha);
  fatomic_add(op + lane + 32, hp[lane + 32] * alpha);
}

// ---------------------------------------------------------------------------
// out = LayerNorm(relu(agg + b)) * gamma + beta   (wave per node)
// ---------------------------------------------------------------------------
__global__ void gat_bias_relu_ln(const float* __restrict__ agg,
                                 const float* __restrict__ b,
                                 const float* __restrict__ gamma,
                                 const float* __restrict__ beta,
                                 float* __restrict__ out, int N) {
  int wv = blockIdx.x * (blockDim.x >> 5) + (threadIdx.x >> 5);
  int lane = threadIdx.x & 31;
  if (wv >= N) return;
  const float* ap = agg + (size_t)wv * 64;
  float v0 = fmaxf(ap[lane]      + b[lane],      0.f);
  float v1 = fmaxf(ap[lane + 32] + b[lane + 32], 0.f);
  float s = v0 + v1;
#pragma unroll
  for (int m = 16; m; m >>= 1) s += __shfl_xor(s, m, 32);
  float mu = s * (1.f / 64.f);
  float c0 = v0 - mu, c1 = v1 - mu;
  float vs = c0 * c0 + c1 * c1;
#pragma unroll
  for (int m = 16; m; m >>= 1) vs += __shfl_xor(vs, m, 32);
  float inv = rsqrtf(vs * (1.f / 64.f) + LN_EPS);
  float* op = out + (size_t)wv * 64;
  op[lane]      = c0 * inv * gamma[lane]      + beta[lane];
  op[lane + 32] = c1 * inv * gamma[lane + 32] + beta[lane + 32];
}

// ---------------------------------------------------------------------------
extern "C" void kernel_launch(void* const* d_in, const int* in_sizes, int n_in,
                              void* d_out, int out_size, void* d_ws, size_t ws_size,
                              hipStream_t stream) {
  const float* x   = (const float*)d_in[0];
  const int*   ei  = (const int*)d_in[1];
  const float* W1  = (const float*)d_in[2];
  const float* aS1 = (const float*)d_in[3];
  const float* aD1 = (const float*)d_in[4];
  const float* b1  = (const float*)d_in[5];
  const float* W2  = (const float*)d_in[6];
  const float* aS2 = (const float*)d_in[7];
  const float* aD2 = (const float*)d_in[8];
  const float* b2  = (const float*)d_in[9];
  const float* g1  = (const float*)d_in[10];
  const float* be1 = (const float*)d_in[11];
  const float* g2  = (const float*)d_in[12];
  const float* be2 = (const float*)d_in[13];

  const int H  = in_sizes[3];          // 64
  const int F  = in_sizes[2] / H;      // 128
  const int N  = in_sizes[0] / F;      // 100000
  const int E  = in_sizes[1] / 2;      // 3200000
  const int ET = E + N;                // + self loops

  // Workspace: 3*N*H + 4*N floats (~78.4 MB)
  float* h   = (float*)d_ws;
  float* hn  = h   + (size_t)N * H;
  float* agg = hn  + (size_t)N * H;
  float* sS  = agg + (size_t)N * H;
  float* sD  = sS + N;
  float* mx  = sD + N;
  float* dn  = mx + N;

  const int TB  = 256;                 // 8 waves per block (wave32)
  const int WPB = TB / 32;
  const int tilesM = (N + 15) / 16;
  dim3 gGemm ((tilesM + WPB - 1) / WPB);
  dim3 gNodeW((N + WPB - 1) / WPB);
  dim3 gEdgeW((ET + WPB - 1) / WPB);
  dim3 gEdgeT((ET + TB - 1) / TB);
  dim3 gReset((N * 64 + TB - 1) / TB);

  // ----- Layer 1 -----
  gat_gemm_wmma     <<<gGemm,  TB, 0, stream>>>(x, W1, h, N, F);
  gat_node_dots     <<<gNodeW, TB, 0, stream>>>(h, aS1, aD1, sS, sD, N);
  gat_reset         <<<gReset, TB, 0, stream>>>(agg, mx, dn, N);
  gat_edge_max      <<<gEdgeT, TB, 0, stream>>>(ei, sS, sD, mx, E, ET);
  gat_edge_expsum   <<<gEdgeT, TB, 0, stream>>>(ei, sS, sD, mx, dn, E, ET);
  gat_edge_aggregate<<<gEdgeW, TB, 0, stream>>>(ei, sS, sD, mx, dn, h, agg, E, ET);
  gat_bias_relu_ln  <<<gNodeW, TB, 0, stream>>>(agg, b1, g1, be1, hn, N);

  // ----- Layer 2 -----
  gat_gemm_wmma     <<<gGemm,  TB, 0, stream>>>(hn, W2, h, N, H);
  gat_node_dots     <<<gNodeW, TB, 0, stream>>>(h, aS2, aD2, sS, sD, N);
  gat_reset         <<<gReset, TB, 0, stream>>>(agg, mx, dn, N);
  gat_edge_max      <<<gEdgeT, TB, 0, stream>>>(ei, sS, sD, mx, E, ET);
  gat_edge_expsum   <<<gEdgeT, TB, 0, stream>>>(ei, sS, sD, mx, dn, E, ET);
  gat_edge_aggregate<<<gEdgeW, TB, 0, stream>>>(ei, sS, sD, mx, dn, h, agg, E, ET);
  gat_bias_relu_ln  <<<gNodeW, TB, 0, stream>>>(agg, b2, g2, be2, (float*)d_out, N);
}
